// LinearAttentionNLP_46574625357881
// MI455X (gfx1250) — matrix operand order repaired and checked by reading, not compiled, not run on hardware
//
#include <hip/hip_runtime.h>

typedef unsigned short u16;
typedef __attribute__((ext_vector_type(16))) __bf16 v16bf;
typedef __attribute__((ext_vector_type(8)))  float  v8f;
typedef __attribute__((ext_vector_type(4)))  unsigned int u32x4;
typedef __attribute__((ext_vector_type(8)))  unsigned int u32x8;

#define DIMC   1024
#define NHEADS 16
#define HDIM   64
#define BATCH  4
#define SEQ    4096
#define MROWS  (BATCH * SEQ)   // 16384

struct alignas(16) U128 { unsigned int x, y, z, w; };
union alignas(32) Frag { v16bf v; U128 q[2]; u16 h[16]; };

__device__ __forceinline__ u16 f2bf(float f) {
  unsigned u = __float_as_uint(f);
  u += 0x7fffu + ((u >> 16) & 1u);          // round-to-nearest-even
  return (u16)(u >> 16);
}
__device__ __forceinline__ float bf2f(u16 h) {
  return __uint_as_float(((unsigned)h) << 16);
}

// A-fragment (16x32 bf16, MxK) from row-major u16 storage (LDS or global).
// Per ISA layout: lanes 0-15 hold K{0-7,16-23}; lanes 16-31 hold K{8-15,24-31}.
__device__ __forceinline__ v16bf load_frag_a(const u16* p, int rowStride, int kOff) {
  const int lane = threadIdx.x & 31;
  const u16* row = p + (size_t)(lane & 15) * rowStride + kOff + ((lane >> 4) << 3);
  Frag f;
  f.q[0] = *(const U128*)(row);
  f.q[1] = *(const U128*)(row + 16);
  return f.v;
}
// B-fragment (32x16 bf16, KxN) from N-major (transposed) storage p[n][k]:
// lane holds column n = lane%16, K{0-7,16-23} (lanes 0-15) / K{8-15,24-31} (lanes 16-31).
__device__ __forceinline__ v16bf load_frag_b(const u16* p, int rowStride, int kOff) {
  const int lane = threadIdx.x & 31;
  const u16* row = p + (size_t)(lane & 15) * rowStride + kOff + ((lane >> 4) << 3);
  Frag f;
  f.q[0] = *(const U128*)(row);
  f.q[1] = *(const U128*)(row + 16);
  return f.v;
}

__device__ __forceinline__ v8f wmma_bf16(v16bf a, v16bf b, v8f c) {
  return __builtin_amdgcn_wmma_f32_16x16x32_bf16(false, a, false, b, (short)0, c, false, false);
}

// -------------------------------------------------------------------- TDM
// DMA one 128-row x 64-element bf16 tile (row stride = strideElems) from
// global into LDS, padding each 128B row with 16B so the LDS row stride is
// 72 halves (matches the fragment loaders).  D# per CDNA5 ISA ch.8:
//   group0: count=1 | lds_addr | global_addr[56:0] | type=2
//   group1: data_size=2B, pad_enable, pad_interval=32 DW (enc 4),
//           pad_amount=4 DW (enc 3), tensor_dim0=strideElems, tensor_dim1=128,
//           tile_dim0=64, tile_dim1=128, tensor_dim0_stride=strideElems
__device__ __forceinline__ void tdm_load_tile(const u16* gsrc, unsigned ldsAddr,
                                              int strideElems) {
  const unsigned long long ga = (unsigned long long)(size_t)gsrc;
  u32x4 g0;
  g0[0] = 1u;                                            // count=1, user mode
  g0[1] = ldsAddr;                                       // LDS byte address
  g0[2] = (unsigned)ga;                                  // global_addr[31:0]
  g0[3] = (unsigned)((ga >> 32) & 0x01FFFFFFull) | (2u << 30);  // [56:32]|type=2
  u32x8 g1;
  g1[0] = (1u << 16) | (1u << 20) | (4u << 22) | (3u << 25);
  g1[1] = ((unsigned)strideElems & 0xFFFFu) << 16;       // tensor_dim0 lo16
  g1[2] = ((unsigned)strideElems >> 16) | (128u << 16);  // dim0 hi | tensor_dim1
  g1[3] = (64u << 16);                                   // dim1 hi=0 | tile_dim0
  g1[4] = 128u;                                          // tile_dim1 | tile_dim2=0
  g1[5] = (unsigned)strideElems;                         // dim0_stride lo32
  g1[6] = 0u;                                            // dim0_stride hi | dim1_stride lo
  g1[7] = 0u;
  asm volatile("tensor_load_to_lds %0, %1" :: "s"(g0), "s"(g1) : "memory");
}

// ---------------------------------------------------------------- prep kernels
__global__ void addpos_kernel(const float* __restrict__ x,
                              const float* __restrict__ pos,
                              u16* __restrict__ xb) {
  const size_t i = ((size_t)blockIdx.x * blockDim.x + threadIdx.x) * 4;
  const float4 xv = *(const float4*)(x + i);
  const float4 pv = *(const float4*)(pos + (i & (size_t)(SEQ * DIMC - 1)));
  struct alignas(8) H4 { u16 h[4]; } o;
  o.h[0] = f2bf(xv.x + pv.x);
  o.h[1] = f2bf(xv.y + pv.y);
  o.h[2] = f2bf(xv.z + pv.z);
  o.h[3] = f2bf(xv.w + pv.w);
  *(H4*)(xb + i) = o;
}

__global__ void cvt_bf16_kernel(const float* __restrict__ src, u16* __restrict__ dst) {
  const size_t i = ((size_t)blockIdx.x * blockDim.x + threadIdx.x) * 4;
  const float4 v = *(const float4*)(src + i);
  struct alignas(8) H4 { u16 h[4]; } o;
  o.h[0] = f2bf(v.x); o.h[1] = f2bf(v.y); o.h[2] = f2bf(v.z); o.h[3] = f2bf(v.w);
  *(H4*)(dst + i) = o;
}

// ---------------------------------------------------------------- tiled GEMM
// C[M x Nout] = A[M x K] * W[Nout x K]^T + bias.
// 128x128 block, BK=64, double-buffered LDS filled by the Tensor Data Mover:
// wave 0 issues two tensor_load_to_lds per K-step (A tile + B tile into the
// other buffer), waits TENSORcnt, and a single barrier per step publishes the
// buffer.  All 8 waves spend the step purely on ds_load_b128 fragments + WMMA.
// MODE 0: QKV epilogue (relu on q/k, scatter to (B,H,N,D) bf16)
// MODE 1: proj epilogue (f32 row-major to d_out)
template <int MODE>
__global__ __launch_bounds__(256) void gemm_bf16_kernel(
    const u16* __restrict__ A, const u16* __restrict__ W,
    const float* __restrict__ bias,
    u16* __restrict__ outQ, u16* __restrict__ outK, u16* __restrict__ outV,
    float* __restrict__ outP, int Kdim) {
  constexpr int BK = 64, PAD = 8, LDA = BK + PAD;
  __shared__ alignas(16) u16 As[2][128][LDA];
  __shared__ alignas(16) u16 Bs[2][128][LDA];

  const int tid  = threadIdx.x;
  const int lane = tid & 31;
  const int wave = tid >> 5;
  const int bm = blockIdx.x * 128;
  const int bn = blockIdx.y * 128;
  const int wm = (wave >> 2) * 64;   // wave grid 2(M) x 4(N)
  const int wn = (wave & 3) * 32;

  v8f acc[4][2] = {};

  const u16* aTile = A + (size_t)bm * Kdim;
  const u16* bTile = W + (size_t)bn * Kdim;

  if (wave == 0) {                       // prologue: DMA tile 0 into buffer 0
    tdm_load_tile(aTile, (unsigned)(size_t)&As[0][0][0], Kdim);
    tdm_load_tile(bTile, (unsigned)(size_t)&Bs[0][0][0], Kdim);
  }

  int buf = 0;
  for (int k0 = 0; k0 < Kdim; k0 += BK) {
    if (wave == 0) __builtin_amdgcn_s_wait_tensorcnt(0);  // buf writes done
    __syncthreads();                                      // publish buf
    if ((k0 + BK) < Kdim && wave == 0) {  // DMA next tile into buf^1; overlaps
      tdm_load_tile(aTile + k0 + BK, (unsigned)(size_t)&As[buf ^ 1][0][0], Kdim);
      tdm_load_tile(bTile + k0 + BK, (unsigned)(size_t)&Bs[buf ^ 1][0][0], Kdim);
    }
#pragma unroll
    for (int kk = 0; kk < BK; kk += 32) {  // 2 WMMA K-steps per staged tile
      const v16bf bf0 = load_frag_b(&Bs[buf][wn][0],      LDA, kk);
      const v16bf bf1 = load_frag_b(&Bs[buf][wn + 16][0], LDA, kk);
#pragma unroll
      for (int i = 0; i < 4; ++i) {
        const v16bf af = load_frag_a(&As[buf][wm + i * 16][0], LDA, kk);
        acc[i][0] = wmma_bf16(af, bf0, acc[i][0]);
        acc[i][1] = wmma_bf16(af, bf1, acc[i][1]);
      }
    }
    buf ^= 1;
  }

#pragma unroll
  for (int j = 0; j < 2; ++j) {
    const int col = bn + wn + j * 16 + (lane & 15);
    const float bv = bias[col];
#pragma unroll
    for (int i = 0; i < 4; ++i) {
#pragma unroll
      for (int e = 0; e < 8; ++e) {
        const int row = bm + wm + i * 16 + e + ((lane >> 4) << 3);
        float v = acc[i][j][e] + bv;
        if constexpr (MODE == 0) {
          const int seg = col >> 10;         // 0=q, 1=k, 2=v
          const int c   = col & 1023;
          const int h   = c >> 6;
          const int d   = c & 63;
          if (seg < 2) v = fmaxf(v, 0.f);    // relu feature map
          const int b = row >> 12;
          const int n = row & 4095;
          const size_t idx = ((size_t)((b << 4) + h) * SEQ + n) * HDIM + d;
          u16* dst = (seg == 0) ? outQ : (seg == 1) ? outK : outV;
          dst[idx] = f2bf(v);
        } else {
          outP[(size_t)row * DIMC + col] = v;
        }
      }
    }
  }
}

// ---------------------------------------------------------------- k_sum
__global__ __launch_bounds__(256) void ksum_kernel(const u16* __restrict__ kb,
                                                   float* __restrict__ ksum) {
  __shared__ float red[256];
  const int bh = blockIdx.x;
  const int tid = threadIdx.x;
  const int d = tid & 63;
  const int part = tid >> 6;
  const u16* kp = kb + (size_t)bh * SEQ * HDIM;
  float s = 0.f;
  for (int n = part; n < SEQ; n += 4) s += bf2f(kp[(size_t)n * HDIM + d]);
  red[tid] = s;
  __syncthreads();
  if (part == 0)
    ksum[bh * HDIM + d] = red[d] + red[64 + d] + red[128 + d] + red[192 + d];
}

// ---------------------------------------------------------------- kv = k^T v
// Per (b,h): 64x64 output, contraction over 4096 rows (split over grid.y=8, 512 each).
// k,v chunks transposed into LDS at staging time so fragments are b128 reads.
__global__ __launch_bounds__(256) void kv_kernel(const u16* __restrict__ kb,
                                                 const u16* __restrict__ vb,
                                                 float* __restrict__ kvpart) {
  __shared__ u16 kt[64][72];   // [c][m]
  __shared__ u16 vt[64][72];   // [d][m]
  const int bh = blockIdx.x;
  const int split = blockIdx.y;
  const int tid = threadIdx.x;
  const int lane = tid & 31;
  const int wave = tid >> 5;
  const int rt  = wave & 3;          // c tile
  const int ct0 = (wave >> 2) * 2;   // 2 d tiles
  const int m  = tid >> 2;
  const int cc = (tid & 3) * 16;

  v8f acc0 = {}, acc1 = {};
  const size_t base = (size_t)bh * SEQ * HDIM;

  for (int mc = split * 512; mc < split * 512 + 512; mc += 64) {
    const u16* kr = kb + base + (size_t)(mc + m) * HDIM + cc;
    const u16* vr = vb + base + (size_t)(mc + m) * HDIM + cc;
    Frag fk, fv;
    fk.q[0] = *(const U128*)(kr); fk.q[1] = *(const U128*)(kr + 8);
    fv.q[0] = *(const U128*)(vr); fv.q[1] = *(const U128*)(vr + 8);
    __syncthreads();
#pragma unroll
    for (int j = 0; j < 16; ++j) {
      kt[cc + j][m] = fk.h[j];
      vt[cc + j][m] = fv.h[j];
    }
    __syncthreads();
#pragma unroll
    for (int ks = 0; ks < 2; ++ks) {
      const v16bf af = load_frag_a(&kt[rt * 16][0], 72, ks * 32);
      const v16bf b0 = load_frag_b(&vt[ct0 * 16][0],       72, ks * 32);
      const v16bf b1 = load_frag_b(&vt[(ct0 + 1) * 16][0], 72, ks * 32);
      acc0 = wmma_bf16(af, b0, acc0);
      acc1 = wmma_bf16(af, b1, acc1);
    }
  }
  float* outp = kvpart + ((size_t)split * 64 + bh) * 4096;
#pragma unroll
  for (int e = 0; e < 8; ++e) {
    const int c = rt * 16 + e + ((lane >> 4) << 3);
    outp[c * 64 + ct0 * 16 + (lane & 15)]       = acc0[e];
    outp[c * 64 + (ct0 + 1) * 16 + (lane & 15)] = acc1[e];
  }
}

// fixed-order partial reduction -> deterministic (no float atomics)
__global__ void kvreduce_kernel(const float* __restrict__ kvpart,
                                u16* __restrict__ kvb) {
  const int i = blockIdx.x * 256 + threadIdx.x;   // 64*4096 elements
  const int bh = i >> 12;
  const int e  = i & 4095;
  float s = 0.f;
#pragma unroll
  for (int p = 0; p < 8; ++p) s += kvpart[((size_t)p * 64 + bh) * 4096 + e];
  kvb[(size_t)bh * 4096 + e] = f2bf(s);
}

// ---------------------------------------------------------------- out = q*kv*z
__global__ __launch_bounds__(256) void apply_kernel(const u16* __restrict__ qb,
                                                    const u16* __restrict__ kvb,
                                                    const float* __restrict__ ksum,
                                                    u16* __restrict__ attn) {
  __shared__ u16 kvt[64][72];   // [d][c]
  __shared__ float ksl[64];
  const int bh = blockIdx.y;
  const int b = bh >> 4;
  const int h = bh & 15;
  const int n0 = blockIdx.x * 128;
  const int tid = threadIdx.x;
  const int lane = tid & 31;
  const int wave = tid >> 5;

  {
    const int c = tid >> 2;
    const int dc = (tid & 3) * 16;
    const u16* src = kvb + (size_t)bh * 4096 + c * 64 + dc;
    Frag f;
    f.q[0] = *(const U128*)(src);
    f.q[1] = *(const U128*)(src + 8);
#pragma unroll
    for (int j = 0; j < 16; ++j) kvt[dc + j][c] = f.h[j];
    if (tid < 64) ksl[tid] = ksum[bh * 64 + tid];
  }
  __syncthreads();

  // A fragments straight from global: q rows are 128B contiguous
  const u16* qrow = qb + ((size_t)bh * SEQ + n0 + wave * 16) * HDIM;
  const v16bf af0 = load_frag_a(qrow, HDIM, 0);
  const v16bf af1 = load_frag_a(qrow, HDIM, 32);

  v8f acc[4] = {};
#pragma unroll
  for (int j = 0; j < 4; ++j) {
    const v16bf b0 = load_frag_b(&kvt[j * 16][0], 72, 0);
    const v16bf b1 = load_frag_b(&kvt[j * 16][0], 72, 32);
    acc[j] = wmma_bf16(af0, b0, acc[j]);
    acc[j] = wmma_bf16(af1, b1, acc[j]);
  }

  // z = 1/(q . ksum + 1e-6) for row (lane&15) of this wave's tile
  const u16* qr = qb + ((size_t)bh * SEQ + n0 + wave * 16 + (lane & 15)) * HDIM;
  float dot = 0.f;
#pragma unroll
  for (int c = 0; c < 64; ++c) dot += bf2f(qr[c]) * ksl[c];
  const float z = 1.f / (dot + 1e-6f);

  u16* outBase = attn + ((size_t)b * SEQ) * DIMC + (size_t)h * HDIM;
#pragma unroll
  for (int e = 0; e < 8; ++e) {
    const int mrow = e + ((lane >> 4) << 3);   // accumulator row 0..15
    const float zm = __int_as_float(
        __builtin_amdgcn_ds_bpermute(mrow << 2, __float_as_int(z)));
    const int nrow = n0 + wave * 16 + mrow;
#pragma unroll
    for (int j = 0; j < 4; ++j)
      outBase[(size_t)nrow * DIMC + j * 16 + (lane & 15)] = f2bf(acc[j][e] * zm);
  }
}

// ---------------------------------------------------------------- launch
extern "C" void kernel_launch(void* const* d_in, const int* in_sizes, int n_in,
                              void* d_out, int out_size, void* d_ws, size_t ws_size,
                              hipStream_t stream) {
  (void)in_sizes; (void)n_in; (void)out_size; (void)ws_size;
  const float* x      = (const float*)d_in[0];
  const float* pos    = (const float*)d_in[1];
  const float* qkv_w  = (const float*)d_in[2];
  const float* qkv_b  = (const float*)d_in[3];
  const float* proj_w = (const float*)d_in[4];
  const float* proj_b = (const float*)d_in[5];
  float* out = (float*)d_out;

  char* ws = (char*)d_ws;
  size_t off = 0;
  auto carve = [&](size_t bytes) {
    void* p = ws + off;
    off = (off + bytes + 255) & ~(size_t)255;
    return p;
  };
  u16*   xb     = (u16*)carve((size_t)MROWS * DIMC * 2);        // 32 MB
  u16*   wqkv   = (u16*)carve((size_t)3 * DIMC * DIMC * 2);     //  6 MB
  u16*   wproj  = (u16*)carve((size_t)DIMC * DIMC * 2);         //  2 MB
  u16*   qb     = (u16*)carve((size_t)MROWS * DIMC * 2);        // 32 MB
  u16*   kb     = (u16*)carve((size_t)MROWS * DIMC * 2);        // 32 MB
  u16*   vb     = (u16*)carve((size_t)MROWS * DIMC * 2);        // 32 MB
  u16*   attn   = (u16*)carve((size_t)MROWS * DIMC * 2);        // 32 MB
  float* ksum   = (float*)carve((size_t)BATCH * NHEADS * HDIM * 4);
  u16*   kvb    = (u16*)carve((size_t)BATCH * NHEADS * HDIM * HDIM * 2);
  float* kvpart = (float*)carve((size_t)8 * BATCH * NHEADS * HDIM * HDIM * 4);

  addpos_kernel<<<dim3((MROWS * DIMC / 4) / 256), 256, 0, stream>>>(x, pos, xb);
  cvt_bf16_kernel<<<dim3((3 * DIMC * DIMC / 4) / 256), 256, 0, stream>>>(qkv_w, wqkv);
  cvt_bf16_kernel<<<dim3((DIMC * DIMC / 4) / 256), 256, 0, stream>>>(proj_w, wproj);

  gemm_bf16_kernel<0><<<dim3(MROWS / 128, (3 * DIMC) / 128), 256, 0, stream>>>(
      xb, wqkv, qkv_b, qb, kb, vb, nullptr, DIMC);

  ksum_kernel<<<dim3(BATCH * NHEADS), 256, 0, stream>>>(kb, ksum);
  kv_kernel<<<dim3(BATCH * NHEADS, 8), 256, 0, stream>>>(kb, vb, kvpart);
  kvreduce_kernel<<<dim3((BATCH * NHEADS * HDIM * HDIM) / 256), 256, 0, stream>>>(kvpart, kvb);
  apply_kernel<<<dim3(SEQ / 128, BATCH * NHEADS), 256, 0, stream>>>(qb, kvb, ksum, attn);

  gemm_bf16_kernel<1><<<dim3(MROWS / 128, DIMC / 128), 256, 0, stream>>>(
      attn, wproj, proj_b, nullptr, nullptr, nullptr, out, DIMC);
}
